// TopPGatingMoE_80350248174109
// MI455X (gfx1250) — compile-verified, matrix operand order checked by reading, not run on hardware
//
#include <hip/hip_runtime.h>
#include <hip/hip_bf16.h>

// ---------------------------------------------------------------------------
// Top-p gated MoE, dense dispatch, for MI455X (gfx1250, wave32, WMMA).
// Heavy math: fp16 WMMA (v_wmma_f32_16x16x32_f16) with fp32 accumulation.
// Weights pre-transposed (fp32->fp16) so both GEMM operands stage into LDS
// with 16B vector ops. Double-buffered LDS, one barrier per K-step,
// 64x64 tile per wave (16 WMMA per K-step per wave).
// ---------------------------------------------------------------------------

typedef _Float16 half_t;
typedef __attribute__((ext_vector_type(16))) _Float16 v16h;
typedef __attribute__((ext_vector_type(8)))  _Float16 v8h;
typedef __attribute__((ext_vector_type(4)))  _Float16 v4h;
typedef __attribute__((ext_vector_type(8)))  float    v8f;

#define BM 128
#define BN 128
#define BK 32
#define LDSK 40            // BK + 8 pad: keeps 16B alignment, staggers banks
#define LDS_TILE (BM * LDSK)

#define NTOK 4096          // B*T
#define DIM_H 1024
#define DIM_F 4096
#define NEXP 8
#define TOPP 0.8f

union Frag { v16h v; v8h h[2]; };

// Fragment load per CDNA5 16-bit A-matrix layout (ISA 7.12.2):
// lane<16 : row=lane,    K {0..7} then {16..23}
// lane>=16: row=lane-16, K {8..15} then {24..31}
__device__ __forceinline__ v16h load_frag_lds(const half_t* base, int lrow, int khalf) {
  Frag f;
  const half_t* p = base + lrow * LDSK + khalf;
  f.h[0] = *reinterpret_cast<const v8h*>(p);
  f.h[1] = *reinterpret_cast<const v8h*>(p + 16);
  return f.v;
}

// Stage a 128x32 fp16 tile (row-major, leading dim = ldk) into registers.
// 128 threads * 4 chunks of 8 halves = 4096 elements.
__device__ __forceinline__ void stage_load(const half_t* __restrict__ src, int ldk,
                                           int r0, int k0, int tid, v8h r[4]) {
#pragma unroll
  for (int it = 0; it < 4; ++it) {
    const int idx = it * 128 + tid;       // 0..511
    const int row = idx >> 2;             // 0..127
    const int col = (idx & 3) << 3;       // 0,8,16,24
    r[it] = *reinterpret_cast<const v8h*>(src + (size_t)(r0 + row) * ldk + k0 + col);
  }
}

__device__ __forceinline__ void stage_store(half_t* dst, int tid, const v8h r[4]) {
#pragma unroll
  for (int it = 0; it < 4; ++it) {
    const int idx = it * 128 + tid;
    const int row = idx >> 2;
    const int col = (idx & 3) << 3;
    *reinterpret_cast<v8h*>(&dst[row * LDSK + col]) = r[it];
  }
}

// Core: acc[4][4] += A_tile(BMxK) * B_tile(KxBN).
// A:  row-major (M,K), lda=K.  BT: row-major (N,K) (pre-transposed weights).
// 128 threads = 4 waves in 2x2 grid; each wave computes 64x64.
__device__ __forceinline__ void gemm_core(const half_t* __restrict__ A,
                                          const half_t* __restrict__ BT,
                                          int K, int m0, int n0,
                                          half_t* lds_a, half_t* lds_b,  // 2*LDS_TILE each
                                          v8f acc[4][4]) {
  const int tid   = threadIdx.x;
  const int lane  = tid & 31;
  const int wid   = tid >> 5;
  const int waveM = wid & 1;           // 2 waves along M
  const int waveN = wid >> 1;          // 2 waves along N
  const int lrow  = lane & 15;
  const int khalf = (lane >> 4) << 3;  // 0 or 8

  v8h ra[4], rb[4];
  stage_load(A,  K, m0, 0, tid, ra);
  stage_load(BT, K, n0, 0, tid, rb);
  stage_store(lds_a, tid, ra);
  stage_store(lds_b, tid, rb);
  __syncthreads();

  int cur = 0;
  for (int k0 = 0; k0 < K; k0 += BK) {
    const bool has_next = (k0 + BK) < K;
    if (has_next) {                       // prefetch next tile (hides VMEM latency)
      stage_load(A,  K, m0, k0 + BK, tid, ra);
      stage_load(BT, K, n0, k0 + BK, tid, rb);
    }
    const half_t* la = lds_a + cur * LDS_TILE;
    const half_t* lb = lds_b + cur * LDS_TILE;
    v16h af[4], bf[4];
#pragma unroll
    for (int mi = 0; mi < 4; ++mi)
      af[mi] = load_frag_lds(la + (waveM * 64 + mi * 16) * LDSK, lrow, khalf);
#pragma unroll
    for (int ni = 0; ni < 4; ++ni)
      bf[ni] = load_frag_lds(lb + (waveN * 64 + ni * 16) * LDSK, lrow, khalf);
#pragma unroll
    for (int mi = 0; mi < 4; ++mi)
#pragma unroll
      for (int ni = 0; ni < 4; ++ni)
        acc[mi][ni] = __builtin_amdgcn_wmma_f32_16x16x32_f16(
            false, af[mi], false, bf[ni], (short)0, acc[mi][ni], false, false);
    if (has_next) {                       // write into the *other* buffer: no WAR
      stage_store(lds_a + (cur ^ 1) * LDS_TILE, tid, ra);
      stage_store(lds_b + (cur ^ 1) * LDS_TILE, tid, rb);
      cur ^= 1;
    }
    __syncthreads();                      // single barrier per K-step
  }
}

// h = relu(x @ W1[e] + b1[e]) -> fp16
__global__ __launch_bounds__(128)
void moe_ffn1(const half_t* __restrict__ A, const half_t* __restrict__ BT,
              const float* __restrict__ bias, half_t* __restrict__ Hout,
              int K, int N) {
  __shared__ half_t lds_a[2 * LDS_TILE];
  __shared__ half_t lds_b[2 * LDS_TILE];
  const int m0 = blockIdx.x * BM;
  const int n0 = blockIdx.y * BN;

  v8f acc[4][4];
  const v8f vz = {};
#pragma unroll
  for (int i = 0; i < 4; ++i)
#pragma unroll
    for (int j = 0; j < 4; ++j) acc[i][j] = vz;

  gemm_core(A, BT, K, m0, n0, lds_a, lds_b, acc);

  const int lane  = threadIdx.x & 31;
  const int wid   = threadIdx.x >> 5;
  const int waveM = wid & 1, waveN = wid >> 1;
  const int nlane = lane & 15;
  const int mhalf = (lane >> 4) << 3;
#pragma unroll
  for (int ni = 0; ni < 4; ++ni) {
    const int n  = n0 + waveN * 64 + ni * 16 + nlane;
    const float bv = bias[n];
#pragma unroll
    for (int mi = 0; mi < 4; ++mi) {
#pragma unroll
      for (int r = 0; r < 8; ++r) {
        const int m = m0 + waveM * 64 + mi * 16 + mhalf + r;
        float v = acc[mi][ni][r] + bv;
        v = v > 0.f ? v : 0.f;
        Hout[(size_t)m * N + n] = (half_t)v;
      }
    }
  }
}

// out += gate[:,e] * (h @ W2[e] + b2[e])
__global__ __launch_bounds__(128)
void moe_ffn2(const half_t* __restrict__ A, const half_t* __restrict__ BT,
              const float* __restrict__ bias, const float* __restrict__ gate,
              int expert, float* __restrict__ Out, int K, int N) {
  __shared__ half_t lds_a[2 * LDS_TILE];
  __shared__ half_t lds_b[2 * LDS_TILE];
  __shared__ float  s_gate[BM];
  const int m0 = blockIdx.x * BM;
  const int n0 = blockIdx.y * BN;

  if (threadIdx.x < BM)
    s_gate[threadIdx.x] = gate[(size_t)(m0 + threadIdx.x) * NEXP + expert];

  v8f acc[4][4];
  const v8f vz = {};
#pragma unroll
  for (int i = 0; i < 4; ++i)
#pragma unroll
    for (int j = 0; j < 4; ++j) acc[i][j] = vz;

  gemm_core(A, BT, K, m0, n0, lds_a, lds_b, acc);  // contains barriers

  const int lane  = threadIdx.x & 31;
  const int wid   = threadIdx.x >> 5;
  const int waveM = wid & 1, waveN = wid >> 1;
  const int nlane = lane & 15;
  const int mhalf = (lane >> 4) << 3;
#pragma unroll
  for (int ni = 0; ni < 4; ++ni) {
    const int n  = n0 + waveN * 64 + ni * 16 + nlane;
    const float bv = bias[n];
#pragma unroll
    for (int mi = 0; mi < 4; ++mi) {
#pragma unroll
      for (int r = 0; r < 8; ++r) {
        const int mloc = waveM * 64 + mi * 16 + mhalf + r;
        const int m    = m0 + mloc;
        Out[(size_t)m * N + n] += s_gate[mloc] * (acc[mi][ni][r] + bv);
      }
    }
  }
}

// Router: one wave per token. probs=softmax(x@Wr+br); descending sort (not stable);
// inclusive cumsum; keep cum<TOP_P (top-1 forced); gate = probs*keep. All fp32.
__global__ __launch_bounds__(256)
void moe_router(const float* __restrict__ x, const float* __restrict__ Wr,
                const float* __restrict__ br, float* __restrict__ gate) {
  const int lane = threadIdx.x & 31;
  const int wid  = threadIdx.x >> 5;
  const int t    = blockIdx.x * 8 + wid;

  float acc[NEXP];
#pragma unroll
  for (int e = 0; e < NEXP; ++e) acc[e] = 0.f;

  const float* xr = x + (size_t)t * DIM_H;
  for (int i = lane; i < DIM_H; i += 32) {
    const float xv = xr[i];
    const float* wr = Wr + (size_t)i * NEXP;
#pragma unroll
    for (int e = 0; e < NEXP; ++e) acc[e] += xv * wr[e];
  }
#pragma unroll
  for (int e = 0; e < NEXP; ++e) {
#pragma unroll
    for (int off = 16; off > 0; off >>= 1)
      acc[e] += __shfl_xor(acc[e], off, 32);
  }

  if (lane == 0) {
    float p[NEXP];
    float mx = -1e30f;
#pragma unroll
    for (int e = 0; e < NEXP; ++e) { p[e] = acc[e] + br[e]; mx = p[e] > mx ? p[e] : mx; }
    float sum = 0.f;
#pragma unroll
    for (int e = 0; e < NEXP; ++e) { p[e] = expf(p[e] - mx); sum += p[e]; }
    const float inv = 1.f / sum;
#pragma unroll
    for (int e = 0; e < NEXP; ++e) p[e] *= inv;

    int idx[NEXP];
#pragma unroll
    for (int e = 0; e < NEXP; ++e) idx[e] = e;
    for (int i = 0; i < NEXP - 1; ++i) {       // descending selection sort (swap-based, not stable)
      int best = i;
      for (int j = i + 1; j < NEXP; ++j)
        if (p[idx[j]] > p[idx[best]]) best = j;
      int tmp = idx[i]; idx[i] = idx[best]; idx[best] = tmp;
    }
    float g[NEXP];
    float cum = 0.f;
    for (int i = 0; i < NEXP; ++i) {
      const float pi = p[idx[i]];
      cum += pi;
      const bool keep = (i == 0) || (cum < TOPP);
      g[idx[i]] = keep ? pi : 0.f;
    }
    float* go = gate + (size_t)t * NEXP;
#pragma unroll
    for (int e = 0; e < NEXP; ++e) go[e] = g[e];
  }
}

// fp32 (K,N) row-major  ->  fp16 (N,K) row-major, 32x32 tiles via LDS.
__global__ __launch_bounds__(256)
void transpose_cvt(const float* __restrict__ in, half_t* __restrict__ out,
                   int K, int N) {
  __shared__ half_t tile[32][40];
  const int col  = threadIdx.x & 31;
  const int rowg = threadIdx.x >> 5;       // 0..7
  const int k0 = blockIdx.x * 32;
  const int n0 = blockIdx.y * 32;
#pragma unroll
  for (int r = 0; r < 4; ++r) {
    const int row = r * 8 + rowg;
    tile[row][col] = (half_t)in[(size_t)(k0 + row) * N + n0 + col];
  }
  __syncthreads();
#pragma unroll
  for (int r = 0; r < 4; ++r) {
    const int row = r * 8 + rowg;          // output n-index within tile
    out[(size_t)(n0 + row) * K + k0 + col] = tile[col][row];
  }
}

__global__ __launch_bounds__(256)
void cvt_f32_to_f16(const float* __restrict__ in, half_t* __restrict__ out, int n) {
  int i = (blockIdx.x * blockDim.x + threadIdx.x) * 4;
  const int stride = gridDim.x * blockDim.x * 4;
  for (; i < n; i += stride) {
    const float4 v = *reinterpret_cast<const float4*>(in + i);
    v4h o;
    o[0] = (half_t)v.x; o[1] = (half_t)v.y; o[2] = (half_t)v.z; o[3] = (half_t)v.w;
    *reinterpret_cast<v4h*>(out + i) = o;
  }
}

__global__ __launch_bounds__(256)
void zero_f32(float* __restrict__ p, int n) {
  int i = (blockIdx.x * blockDim.x + threadIdx.x) * 4;
  const int stride = gridDim.x * blockDim.x * 4;
  const float4 z = {0.f, 0.f, 0.f, 0.f};
  for (; i < n; i += stride) *reinterpret_cast<float4*>(p + i) = z;
}

extern "C" void kernel_launch(void* const* d_in, const int* in_sizes, int n_in,
                              void* d_out, int out_size, void* d_ws, size_t ws_size,
                              hipStream_t stream) {
  (void)in_sizes; (void)n_in; (void)out_size; (void)ws_size;

  const float* x  = (const float*)d_in[0];   // (B,T,H)
  const float* Wr = (const float*)d_in[1];   // (H,E)
  const float* br = (const float*)d_in[2];   // (E)
  const float* W1 = (const float*)d_in[3];   // (E,H,F)
  const float* b1 = (const float*)d_in[4];   // (E,F)
  const float* W2 = (const float*)d_in[5];   // (E,F,H)
  const float* b2 = (const float*)d_in[6];   // (E,H)
  float* out = (float*)d_out;                // (B,T,H) fp32

  // Workspace layout (all 256B aligned): ~59 MB total
  char* ws = (char*)d_ws;
  float*  gate = (float*)(ws);                               // 128 KB
  half_t* xb   = (half_t*)(ws + (131072));                   // 8 MB
  half_t* w1t  = (half_t*)(ws + (131072 + 8388608));         // 8 MB (F,H) fp16
  half_t* w2t  = (half_t*)(ws + (131072 + 2 * 8388608));     // 8 MB (H,F) fp16
  half_t* hbuf = (half_t*)(ws + (131072 + 3 * 8388608));     // 32 MB (M,F) fp16

  zero_f32<<<1024, 256, 0, stream>>>(out, NTOK * DIM_H);
  moe_router<<<NTOK / 8, 256, 0, stream>>>(x, Wr, br, gate);
  cvt_f32_to_f16<<<2048, 256, 0, stream>>>(x, xb, NTOK * DIM_H);

  const dim3 g1(NTOK / BM, DIM_F / BN);     // 32 x 32
  const dim3 g2(NTOK / BM, DIM_H / BN);     // 32 x 8
  const dim3 t1(DIM_H / 32, DIM_F / 32);    // W1: (H,F) -> (F,H)
  const dim3 t2(DIM_F / 32, DIM_H / 32);    // W2: (F,H) -> (H,F)

  for (int e = 0; e < NEXP; ++e) {
    transpose_cvt<<<t1, 256, 0, stream>>>(W1 + (size_t)e * DIM_H * DIM_F, w1t, DIM_H, DIM_F);
    transpose_cvt<<<t2, 256, 0, stream>>>(W2 + (size_t)e * DIM_F * DIM_H, w2t, DIM_F, DIM_H);
    moe_ffn1<<<g1, 128, 0, stream>>>(xb, w1t, b1 + (size_t)e * DIM_F, hbuf, DIM_H, DIM_F);
    moe_ffn2<<<g2, 128, 0, stream>>>(hbuf, w2t, b2 + (size_t)e * DIM_H, gate, e, out, DIM_F, DIM_H);
  }
}
